// SuperGAT_19782619365934
// MI455X (gfx1250) — matrix operand order repaired and checked by reading, not compiled, not run on hardware
//
#include <hip/hip_runtime.h>
#include <hip/hip_bf16.h>
#include <cstdint>

// SuperGAT 2-layer forward for MI455X (gfx1250, wave32).
// GEMMs use v_wmma_f32_16x16x32_f16 (f16 in, f32 accum). Edge phases use
// coalesced gathers + lane-group shuffle reductions + L2 float atomics for
// the segment softmax / scatter-sum (xh arrays are L2-resident: 25.6/51.2 MB
// vs 192 MB L2). Note: float atomic-add order is hardware-scheduled, so
// results are numerically stable but not bitwise-identical across replays.

typedef __attribute__((ext_vector_type(16))) _Float16 v16h;
typedef __attribute__((ext_vector_type(8)))  _Float16 v8h;
typedef __attribute__((ext_vector_type(8)))  float    v8f;

#define NEG_SLOPE 0.2f

__device__ __forceinline__ unsigned enc_f32(float f) {
  unsigned b = __float_as_uint(f);
  return (b & 0x80000000u) ? ~b : (b | 0x80000000u);   // order-preserving map
}
__device__ __forceinline__ float dec_f32(unsigned u) {
  unsigned b = (u & 0x80000000u) ? (u & 0x7fffffffu) : ~u;
  return __uint_as_float(b);
}
#define ENC_NEG_INF 0x007fffffu   // enc_f32(-inf)

__device__ __forceinline__ v8f wmma16x16x32(v16h a, v16h b, v8f c) {
  // (neg_a, A, neg_b, B, c_mod, C, reuse_a, reuse_b)
  return __builtin_amdgcn_wmma_f32_16x16x32_f16(false, a, false, b, (short)0, c,
                                                false, false);
}

// ---------------------------------------------------------------- utilities
__global__ void fill_u32(unsigned* __restrict__ p, unsigned v, int count) {
  int i = blockIdx.x * blockDim.x + threadIdx.x;
  if (i < count) p[i] = v;
}

// Pack W1 (256x64) and W2 (64x128) f32 row-major into per-lane WMMA B layout:
// Bp[((kb*NT + nt)*32 + lane)*16 + j] = W[(kb*32 + 16*(lane>>4) + j)*NOUT + nt*16 + (lane&15)]
__global__ void pack_w(const float* __restrict__ W1, const float* __restrict__ W2,
                       _Float16* __restrict__ Bp1, _Float16* __restrict__ Bp2) {
  int t = blockIdx.x * blockDim.x + threadIdx.x;
  if (t < 16384) {                       // W1: KB=8, NT=4
    int j = t & 15, lane = (t >> 4) & 31, nt = (t >> 9) & 3, kb = t >> 11;
    int g = lane >> 4, nl = lane & 15;
    int k = kb * 32 + 16 * g + j;
    int nn = nt * 16 + nl;
    Bp1[t] = (_Float16)W1[k * 64 + nn];
  } else if (t < 16384 + 8192) {         // W2: KB=2, NT=8
    int u = t - 16384;
    int j = u & 15, lane = (u >> 4) & 31, nt = (u >> 9) & 7, kb = u >> 12;
    int g = lane >> 4, nl = lane & 15;
    int k = kb * 32 + 16 * g + j;
    int nn = nt * 16 + nl;
    Bp2[u] = (_Float16)W2[k * 128 + nn];
  }
}

// h16[n][256] = (f16) emb[x[n]][256]; 8 floats per thread
__global__ void embed_gather_f16(const int* __restrict__ x, const float* __restrict__ emb,
                                 _Float16* __restrict__ h16, int n) {
  int gid = blockIdx.x * blockDim.x + threadIdx.x;
  if (gid >= n * 32) return;
  int node = gid >> 5, part = gid & 31;
  const float4* s = (const float4*)(emb + (size_t)x[node] * 256 + part * 8);
  float4 a = s[0], b = s[1];
  v8h o = { (_Float16)a.x, (_Float16)a.y, (_Float16)a.z, (_Float16)a.w,
            (_Float16)b.x, (_Float16)b.y, (_Float16)b.z, (_Float16)b.w };
  *(v8h*)(h16 + (size_t)node * 256 + part * 8) = o;
}

// ------------------------------------------------------------- WMMA GEMM
// C[nrows x NOUT] = A[nrows x K] (f16, row major) * B (prepacked f16), f32 out.
// One wave per 16-row tile; full EXEC maintained (guards are wave-uniform).
template <int K, int NOUT>
__global__ void gemm_wmma(const _Float16* __restrict__ A, const _Float16* __restrict__ Bp,
                          float* __restrict__ C, int nrows) {
  constexpr int KB = K / 32;
  constexpr int NT = NOUT / 16;
  int wave = blockIdx.x * (blockDim.x >> 5) + (threadIdx.x >> 5);
  int lane = threadIdx.x & 31;
  int row0 = wave * 16;
  if (row0 >= nrows) return;                    // wave-uniform
  int m = lane & 15, g = lane >> 4;
  int arow = row0 + m; if (arow > nrows - 1) arow = nrows - 1;
  v8f acc[NT];
#pragma unroll
  for (int i = 0; i < NT; ++i) { v8f z = {}; acc[i] = z; }
#pragma unroll
  for (int kb = 0; kb < KB; ++kb) {
    // A layout (contiguous, matches packed B): lane m+16g holds K = 16g..16g+15
    const _Float16* ap = A + (size_t)arow * K + kb * 32 + 16 * g;
    v8h alo = *(const v8h*)ap;
    v8h ahi = *(const v8h*)(ap + 8);
    v16h av = __builtin_shufflevector(alo, ahi, 0, 1, 2, 3, 4, 5, 6, 7,
                                      8, 9, 10, 11, 12, 13, 14, 15);
#pragma unroll
    for (int nt = 0; nt < NT; ++nt) {
      v16h bv = *(const v16h*)(Bp + ((size_t)(kb * NT + nt) * 32 + lane) * 16);
      acc[nt] = wmma16x16x32(av, bv, acc[nt]);
    }
  }
  // D layout: VGPR v, lane n+16g -> element (row0 + v + 8g, nt*16 + n)
  int nlo = lane & 15, gh = lane >> 4;
#pragma unroll
  for (int v = 0; v < 8; ++v) {
    int r = row0 + v + 8 * gh;
    if (r < nrows) {
#pragma unroll
      for (int nt = 0; nt < NT; ++nt)
        C[(size_t)r * NOUT + nt * 16 + nlo] = acc[nt][v];
    }
  }
}

// ------------------------------------------------------------ edge phases
// CH = per-head channels (8 or 16); VEC = 8*CH floats per node vector.
template <int CH>
__global__ void edge_alpha(const float* __restrict__ xh, const int* __restrict__ src,
                           const int* __restrict__ dst, int E, int Etot,
                           const float* __restrict__ att, float* __restrict__ alpha,
                           unsigned* __restrict__ amax) {
  constexpr int VEC = 8 * CH;
  int gid = blockIdx.x * blockDim.x + threadIdx.x;
  int e = gid / VEC;
  if (e >= Etot) return;
  int t = threadIdx.x % VEC;                      // blockDim % VEC == 0
  int s, d;
  if (e < E) { s = src[e]; d = dst[e]; } else { s = e - E; d = s; }
  float xi = xh[(size_t)d * VEC + t];
  float xj = xh[(size_t)s * VEC + t];
  int h = t / CH, c = t % CH;
  float lg = xi * xj;                                    // -> logits[h]
  float av = xi * att[h * 2 * CH + c] + xj * att[h * 2 * CH + CH + c];
#pragma unroll
  for (int off = CH / 2; off > 0; off >>= 1) {           // in-wave group reduce
    lg += __shfl_xor(lg, off, CH);
    av += __shfl_xor(av, off, CH);
  }
  if (c == 0) {
    float sg = 1.f / (1.f + __expf(-lg));
    float al = av * sg;
    al = al > 0.f ? al : NEG_SLOPE * al;                 // leaky relu
    alpha[(size_t)e * 8 + h] = al;
    atomicMax(amax + (size_t)d * 8 + h, enc_f32(al));    // segment max
  }
}

__global__ void edge_softmax(const int* __restrict__ dst, int E, int Etot,
                             float* __restrict__ alpha, const unsigned* __restrict__ amax,
                             float* __restrict__ denom) {
  int i = blockIdx.x * blockDim.x + threadIdx.x;
  if (i >= Etot * 8) return;
  int e = i >> 3, h = i & 7;
  int d = (e < E) ? dst[e] : (e - E);
  float ex = __expf(alpha[i] - dec_f32(amax[(size_t)d * 8 + h]));
  alpha[i] = ex;                                          // in-place: alpha -> ex
  atomicAdd(denom + (size_t)d * 8 + h, ex);               // segment sum
}

template <int CH>
__global__ void edge_agg(const float* __restrict__ xh, const int* __restrict__ src,
                         const int* __restrict__ dst, int E, int Etot,
                         const float* __restrict__ ex, const float* __restrict__ denom,
                         float* __restrict__ out) {
  constexpr int VEC = 8 * CH;
  int gid = blockIdx.x * blockDim.x + threadIdx.x;
  int e = gid / VEC;
  if (e >= Etot) return;
  int t = threadIdx.x % VEC;
  int s, d;
  if (e < E) { s = src[e]; d = dst[e]; } else { s = e - E; d = s; }
  int h = t / CH;
  float a = ex[(size_t)e * 8 + h] / (denom[(size_t)d * 8 + h] + 1e-16f);
  atomicAdd(out + (size_t)d * VEC + t, xh[(size_t)s * VEC + t] * a);
}

// out1 + b1 -> ELU -> f16 (feeds GEMM2)
__global__ void bias_elu_f16(const float* __restrict__ in, const float* __restrict__ bias,
                             _Float16* __restrict__ out, int total, int vw) {
  int i = blockIdx.x * blockDim.x + threadIdx.x;
  if (i >= total) return;
  float v = in[i] + bias[i % vw];
  v = v > 0.f ? v : (__expf(v) - 1.f);
  out[i] = (_Float16)v;
}

// mean over 8 heads, + b2, log_softmax over 16 classes
__global__ void finalize(const float* __restrict__ out2, const float* __restrict__ b2,
                         float* __restrict__ out, int n) {
  int i = blockIdx.x * blockDim.x + threadIdx.x;
  if (i >= n) return;
  float v[16];
#pragma unroll
  for (int c = 0; c < 16; ++c) v[c] = 0.f;
  const float* p = out2 + (size_t)i * 128;
#pragma unroll
  for (int h = 0; h < 8; ++h)
#pragma unroll
    for (int c = 0; c < 16; ++c) v[c] += p[h * 16 + c];
  float mx = -1e30f;
#pragma unroll
  for (int c = 0; c < 16; ++c) { v[c] = v[c] * 0.125f + b2[c]; mx = fmaxf(mx, v[c]); }
  float ssum = 0.f;
#pragma unroll
  for (int c = 0; c < 16; ++c) ssum += __expf(v[c] - mx);
  float lse = mx + __logf(ssum);
#pragma unroll
  for (int c = 0; c < 16; ++c) out[(size_t)i * 16 + c] = v[c] - lse;
}

// ---------------------------------------------------------------- launcher
extern "C" void kernel_launch(void* const* d_in, const int* in_sizes, int n_in,
                              void* d_out, int out_size, void* d_ws, size_t ws_size,
                              hipStream_t stream) {
  (void)n_in; (void)out_size; (void)ws_size;
  const int*   x    = (const int*)d_in[0];
  const int*   ei   = (const int*)d_in[1];
  const float* emb  = (const float*)d_in[2];
  const float* W1   = (const float*)d_in[3];
  const float* att1 = (const float*)d_in[4];
  const float* b1   = (const float*)d_in[5];
  const float* W2   = (const float*)d_in[6];
  const float* att2 = (const float*)d_in[7];
  const float* b2   = (const float*)d_in[8];
  const int n = in_sizes[0];
  const int E = in_sizes[1] / 2;
  const int Etot = E + n;
  const int* src = ei;
  const int* dst = ei + E;

  // workspace carving (256B aligned); h16 region is reused as out2
  // (n*256*2 bytes f16 == n*128*4 bytes f32), alpha buffer reused across layers
  char* ws = (char*)d_ws;
  size_t off = 0;
  auto carve = [&](size_t bytes) {
    size_t o = off; off = (off + bytes + 255) & ~(size_t)255; return o;
  };
  size_t o_h16  = carve((size_t)n * 256 * 2);
  size_t o_xh1  = carve((size_t)n * 64 * 4);
  size_t o_alp  = carve((size_t)Etot * 8 * 4);
  size_t o_amax = carve((size_t)n * 8 * 4);
  size_t o_den  = carve((size_t)n * 8 * 4);
  size_t o_out1 = carve((size_t)n * 64 * 4);
  size_t o_h216 = carve((size_t)n * 64 * 2);
  size_t o_xh2  = carve((size_t)n * 128 * 4);
  size_t o_bp1  = carve(256 * 64 * 2);
  size_t o_bp2  = carve(64 * 128 * 2);

  _Float16* h16  = (_Float16*)(ws + o_h16);
  float*    out2 = (float*)(ws + o_h16);      // alias: h16 dead after GEMM1
  float*    xh1  = (float*)(ws + o_xh1);
  float*    alp  = (float*)(ws + o_alp);
  unsigned* amax = (unsigned*)(ws + o_amax);
  float*    den  = (float*)(ws + o_den);
  float*    out1 = (float*)(ws + o_out1);
  _Float16* h216 = (_Float16*)(ws + o_h216);
  float*    xh2  = (float*)(ws + o_xh2);
  _Float16* bp1  = (_Float16*)(ws + o_bp1);
  _Float16* bp2  = (_Float16*)(ws + o_bp2);

  const int B = 256;
  auto blocks = [](long long t, int b) { return (unsigned)((t + b - 1) / b); };

  // prep
  pack_w<<<blocks(16384 + 8192, B), B, 0, stream>>>(W1, W2, bp1, bp2);
  embed_gather_f16<<<blocks((long long)n * 32, B), B, 0, stream>>>(x, emb, h16, n);

  // ---- layer 1 ----
  fill_u32<<<blocks(n * 8, B), B, 0, stream>>>(amax, ENC_NEG_INF, n * 8);
  fill_u32<<<blocks(n * 8, B), B, 0, stream>>>((unsigned*)den, 0u, n * 8);
  fill_u32<<<blocks(n * 64, B), B, 0, stream>>>((unsigned*)out1, 0u, n * 64);
  const int mtiles = (n + 15) / 16;
  gemm_wmma<256, 64><<<blocks(mtiles, 8), B, 0, stream>>>(h16, bp1, xh1, n);
  edge_alpha<8><<<blocks((long long)Etot * 64, B), B, 0, stream>>>(
      xh1, src, dst, E, Etot, att1, alp, amax);
  edge_softmax<<<blocks((long long)Etot * 8, B), B, 0, stream>>>(
      dst, E, Etot, alp, amax, den);
  edge_agg<8><<<blocks((long long)Etot * 64, B), B, 0, stream>>>(
      xh1, src, dst, E, Etot, alp, den, out1);
  bias_elu_f16<<<blocks((long long)n * 64, B), B, 0, stream>>>(out1, b1, h216, n * 64, 64);

  // ---- layer 2 ----
  fill_u32<<<blocks(n * 8, B), B, 0, stream>>>(amax, ENC_NEG_INF, n * 8);
  fill_u32<<<blocks(n * 8, B), B, 0, stream>>>((unsigned*)den, 0u, n * 8);
  fill_u32<<<blocks(n * 128, B), B, 0, stream>>>((unsigned*)out2, 0u, n * 128);
  gemm_wmma<64, 128><<<blocks(mtiles, 8), B, 0, stream>>>(h216, bp2, xh2, n);
  edge_alpha<16><<<blocks((long long)Etot * 128, B), B, 0, stream>>>(
      xh2, src, dst, E, Etot, att2, alp, amax);
  edge_softmax<<<blocks((long long)Etot * 8, B), B, 0, stream>>>(
      dst, E, Etot, alp, amax, den);
  edge_agg<16><<<blocks((long long)Etot * 128, B), B, 0, stream>>>(
      xh2, src, dst, E, Etot, alp, den, out2);

  finalize<<<blocks(n, B), B, 0, stream>>>(out2, b2, (float*)d_out, n);
}
